// BasicLSTMModel_87196426043838
// MI455X (gfx1250) — matrix operand order repaired and checked by reading, not compiled, not run on hardware
//
#include <hip/hip_runtime.h>
#include <hip/hip_bf16.h>

typedef __bf16  v16bf __attribute__((ext_vector_type(16)));
typedef __bf16  v8bf  __attribute__((ext_vector_type(8)));
typedef float   v8f   __attribute__((ext_vector_type(8)));

#define TSTEPS 512
#define BSZ    256
#define FDIM   64
#define U1N    256
#define U2N    128
#define NWG    16

// LDS layout pads (halves / floats) to dodge bank conflicts
#define LDB1 328   // K=320 padded
#define LDB2 392   // K=384 padded
#define CP1  33    // 32 units padded
#define CP2  17    // 16 units padded

#define SMEM_BYTES (128*LDB1*2 + 256*CP1*4)   // 83968 + 33792 = 117760 (L1 role, the max)

// ---- workspace layout (bytes) ----
// [0,256)          : sync (cnt, gen)
// [256, ...)       : h1buf[2] (2*65536 bf16) then h2buf[2] (2*32768 bf16)
// [393472, ...)    : xbf  [T][B][64] bf16  (16.8 MB)
#define WS_H1_OFF   256
#define WS_H2_OFF   (256 + 2*65536*2)
#define WS_X_OFF    393472
#define WS_ZERO_BYTES 393472

static __device__ __forceinline__ float sigf(float x) { return 1.0f / (1.0f + __expf(-x)); }

// A-fragment: 16x32 bf16, lane holds row (lane&15), K = base+(lanehi?8:0) in halves [0..7],
// and same +16 in halves [8..15]  (ISA 7.12.2)
static __device__ __forceinline__ v16bf ld_afrag(const __bf16* p) {
  union { v16bf v; v8bf h[2]; } u;
  u.h[0] = *(const v8bf*)(p);
  u.h[1] = *(const v8bf*)(p + 16);
  return u.v;
}
// B-fragment from column-major LDS copy: column (lane&15), K = base+(lanehi?8:0) in halves [0..7],
// and same +16 in halves [8..15]
static __device__ __forceinline__ v16bf ld_bfrag(const __bf16* p) {
  union { v16bf v; v8bf h[2]; } u;
  u.h[0] = *(const v8bf*)(p);
  u.h[1] = *(const v8bf*)(p + 16);
  return u.v;
}

static __device__ __forceinline__ void grid_barrier(int* sync) {
  __threadfence();                 // release this thread's stores (agent scope)
  __syncthreads();
  if (threadIdx.x == 0) {
    int g = __hip_atomic_load(&sync[1], __ATOMIC_RELAXED, __HIP_MEMORY_SCOPE_AGENT);
    if (__hip_atomic_fetch_add(&sync[0], 1, __ATOMIC_ACQ_REL, __HIP_MEMORY_SCOPE_AGENT) == NWG - 1) {
      __hip_atomic_store(&sync[0], 0, __ATOMIC_RELAXED, __HIP_MEMORY_SCOPE_AGENT);
      __hip_atomic_fetch_add(&sync[1], 1, __ATOMIC_RELEASE, __HIP_MEMORY_SCOPE_AGENT);
    } else {
      while (__hip_atomic_load(&sync[1], __ATOMIC_ACQUIRE, __HIP_MEMORY_SCOPE_AGENT) == g)
        __builtin_amdgcn_s_sleep(1);
    }
  }
  __syncthreads();
  __threadfence();                 // acquire: invalidate stale cached lines
}

// transpose x [B][T][F] f32 -> [T][B][F] bf16
__global__ void cvt_x_kernel(const float* __restrict__ x, __bf16* __restrict__ xbf) {
  int idx = blockIdx.x * 256 + threadIdx.x;          // t*B*F + b*F + f
  int f = idx & 63;
  int rest = idx >> 6;
  int b = rest & 255;
  int t = rest >> 8;
  xbf[idx] = (__bf16)x[(b * TSTEPS + t) * FDIM + f];
}

__global__ __launch_bounds__(256, 1) void lstm_persistent_kernel(
    const __bf16* __restrict__ xbf,
    const float* __restrict__ W1, const float* __restrict__ U1w, const float* __restrict__ b1,
    const float* __restrict__ W2, const float* __restrict__ U2w, const float* __restrict__ b2,
    __bf16* __restrict__ h1base,   // 2 x [256][256]
    __bf16* __restrict__ h2base,   // 2 x [256][128]
    int* __restrict__ sync) {

  extern __shared__ char smem[];
  const int tid  = threadIdx.x;
  const int wave = tid >> 5;
  const int lane = tid & 31;
  const int lhi  = lane >> 4;      // 0/1
  const int llo  = lane & 15;
  const int wg   = blockIdx.x;
  const bool isL1 = (wg < 8);

  __bf16* ldsB = (__bf16*)smem;
  float*  shc  = isL1 ? (float*)(smem + 128 * LDB1 * 2)
                      : (float*)(smem + 64  * LDB2 * 2);

  float bs1[4][2];   // layer1 gate bias (q, unit-subtile)
  float bs2[4];      // layer2 gate bias
  int u0 = 0, v0 = 0;

  // ---- setup: stage transposed [W;U] slice into LDS, zero c-state ----
  if (isL1) {
    u0 = wg * 32;
    for (int i = tid; i < 128 * 320; i += 256) {
      int col = i / 320, k = i - col * 320;
      int q = col >> 5, ul = col & 31;
      int gc = q * 256 + u0 + ul;                       // global col in [0,1024)
      float v = (k < 64) ? W1[k * 1024 + gc] : U1w[(k - 64) * 1024 + gc];
      ldsB[col * LDB1 + k] = (__bf16)v;
    }
    for (int i = tid; i < 256 * CP1; i += 256) shc[i] = 0.0f;
#pragma unroll
    for (int q = 0; q < 4; ++q)
#pragma unroll
      for (int s = 0; s < 2; ++s)
        bs1[q][s] = b1[q * 256 + u0 + s * 16 + llo];
  } else {
    v0 = (wg - 8) * 16;
    for (int i = tid; i < 64 * 384; i += 256) {
      int col = i / 384, k = i - col * 384;
      int q = col >> 4, ul = col & 15;
      int gc = q * 128 + v0 + ul;                       // global col in [0,512)
      float v = (k < 256) ? W2[k * 512 + gc] : U2w[(k - 256) * 512 + gc];
      ldsB[col * LDB2 + k] = (__bf16)v;
    }
    for (int i = tid; i < 256 * CP2; i += 256) shc[i] = 0.0f;
#pragma unroll
    for (int q = 0; q < 4; ++q) bs2[q] = b2[q * 128 + v0 + llo];
  }
  __syncthreads();

  const int m0 = wave * 2;   // each wave owns M-tiles {m0, m0+1}

  // ---- time loop: L1 computes h1[t] at iter t; L2 consumes h1[t-1] at iter t ----
  for (int t = 0; t <= TSTEPS; ++t) {
    if (isL1) {
      if (t < TSTEPS) {
        const __bf16* hprev = h1base + (t & 1) * (256 * 256);
        __bf16*       hnext = h1base + ((t + 1) & 1) * (256 * 256);
        const __bf16* xt    = xbf + (size_t)t * (BSZ * FDIM);

        // pull next step's x rows toward this WGP while we compute (global_prefetch)
        if (t + 1 < TSTEPS) {
          __builtin_prefetch(xt + (BSZ * FDIM) + (m0 * 16 + llo) * FDIM, 0, 1);
          __builtin_prefetch(xt + (BSZ * FDIM) + ((m0 + 1) * 16 + llo) * FDIM, 0, 1);
        }

        v8f acc[8][2];
#pragma unroll
        for (int n = 0; n < 8; ++n) { acc[n][0] = {}; acc[n][1] = {}; }

#pragma unroll
        for (int kt = 0; kt < 10; ++kt) {
          const __bf16* src; int ld, kb;
          if (kt < 2) { src = xt;    ld = 64;  kb = kt * 32; }
          else        { src = hprev; ld = 256; kb = kt * 32 - 64; }
          const int c0 = kb + lhi * 8;
          v16bf a0 = ld_afrag(src + (m0 * 16 + llo) * ld + c0);
          v16bf a1 = ld_afrag(src + ((m0 + 1) * 16 + llo) * ld + c0);
#pragma unroll
          for (int n = 0; n < 8; ++n) {
            v16bf bf = ld_bfrag(ldsB + (n * 16 + llo) * LDB1 + kt * 32 + lhi * 8);
            acc[n][0] = __builtin_amdgcn_wmma_f32_16x16x32_bf16(false, a0, false, bf, (short)0, acc[n][0], false, false);
            acc[n][1] = __builtin_amdgcn_wmma_f32_16x16x32_bf16(false, a1, false, bf, (short)0, acc[n][1], false, false);
          }
        }
        // gates: n = q*2 + s  (q gate, s unit-subtile of 16)
#pragma unroll
        for (int s = 0; s < 2; ++s) {
          const int ul = s * 16 + llo;
#pragma unroll
          for (int m = 0; m < 2; ++m) {
#pragma unroll
            for (int r = 0; r < 8; ++r) {
              float zi = acc[0 + s][m][r] + bs1[0][s];
              float zf = acc[2 + s][m][r] + bs1[1][s];
              float zg = acc[4 + s][m][r] + bs1[2][s];
              float zo = acc[6 + s][m][r] + bs1[3][s];
              float ig = sigf(zi), fg = sigf(zf), gg = tanhf(zg), og = sigf(zo);
              int b = (m0 + m) * 16 + lhi * 8 + r;
              float* cp = &shc[b * CP1 + ul];
              float cn = fg * (*cp) + ig * gg;
              *cp = cn;
              hnext[b * 256 + u0 + ul] = (__bf16)(og * tanhf(cn));
            }
          }
        }
      }
    } else {
      if (t >= 1) {
        const int s = t - 1;
        const __bf16* h1in  = h1base + (t & 1) * (256 * 256);          // h1[t-1]
        const __bf16* hprev = h2base + (s & 1) * (256 * 128);
        __bf16*       hnext = h2base + ((s + 1) & 1) * (256 * 128);

        v8f acc[4][2];
#pragma unroll
        for (int n = 0; n < 4; ++n) { acc[n][0] = {}; acc[n][1] = {}; }

#pragma unroll
        for (int kt = 0; kt < 12; ++kt) {
          const __bf16* src; int ld, kb;
          if (kt < 8) { src = h1in;  ld = 256; kb = kt * 32; }
          else        { src = hprev; ld = 128; kb = kt * 32 - 256; }
          const int c0 = kb + lhi * 8;
          v16bf a0 = ld_afrag(src + (m0 * 16 + llo) * ld + c0);
          v16bf a1 = ld_afrag(src + ((m0 + 1) * 16 + llo) * ld + c0);
#pragma unroll
          for (int n = 0; n < 4; ++n) {
            v16bf bf = ld_bfrag(ldsB + (n * 16 + llo) * LDB2 + kt * 32 + lhi * 8);
            acc[n][0] = __builtin_amdgcn_wmma_f32_16x16x32_bf16(false, a0, false, bf, (short)0, acc[n][0], false, false);
            acc[n][1] = __builtin_amdgcn_wmma_f32_16x16x32_bf16(false, a1, false, bf, (short)0, acc[n][1], false, false);
          }
        }
#pragma unroll
        for (int m = 0; m < 2; ++m) {
#pragma unroll
          for (int r = 0; r < 8; ++r) {
            float zi = acc[0][m][r] + bs2[0];
            float zf = acc[1][m][r] + bs2[1];
            float zg = acc[2][m][r] + bs2[2];
            float zo = acc[3][m][r] + bs2[3];
            float ig = sigf(zi), fg = sigf(zf), gg = tanhf(zg), og = sigf(zo);
            int b = (m0 + m) * 16 + lhi * 8 + r;
            float* cp = &shc[b * CP2 + llo];
            float cn = fg * (*cp) + ig * gg;
            *cp = cn;
            hnext[b * 128 + v0 + llo] = (__bf16)(og * tanhf(cn));
          }
        }
      }
    }
    grid_barrier(sync);
  }
}

// ---- dense head as its own tiny kernel (keeps scratch out of the persistent kernel) ----
__global__ void head_kernel(
    const __bf16* __restrict__ h2f,         // [256][128] bf16 (final h2, buffer 0)
    const float* __restrict__ Wd1, const float* __restrict__ bd1,
    const float* __restrict__ Wd2, const float* __restrict__ bd2,
    const float* __restrict__ Wo,  const float* __restrict__ bo,
    float* __restrict__ out) {
  const int b = threadIdx.x;
  float h[128];
  for (int k = 0; k < 128; ++k) h[k] = (float)h2f[b * 128 + k];
  float d1[128];
  for (int j = 0; j < 128; ++j) {
    float a = bd1[j];
    for (int k = 0; k < 128; ++k) a += h[k] * Wd1[k * 128 + j];
    d1[j] = a > 0.0f ? a : 0.0f;
  }
  float d2[64];
  for (int j = 0; j < 64; ++j) {
    float a = bd2[j];
    for (int k = 0; k < 128; ++k) a += d1[k] * Wd2[k * 64 + j];
    d2[j] = a > 0.0f ? a : 0.0f;
  }
  float lg[5], mx = -1e30f;
  for (int c = 0; c < 5; ++c) {
    float a = bo[c];
    for (int k = 0; k < 64; ++k) a += d2[k] * Wo[k * 5 + c];
    lg[c] = a;
    mx = a > mx ? a : mx;
  }
  float den = 0.0f;
  for (int c = 0; c < 5; ++c) { lg[c] = __expf(lg[c] - mx); den += lg[c]; }
  float inv = 1.0f / den;
  for (int c = 0; c < 5; ++c) out[b * 5 + c] = lg[c] * inv;
}

extern "C" void kernel_launch(void* const* d_in, const int* in_sizes, int n_in,
                              void* d_out, int out_size, void* d_ws, size_t ws_size,
                              hipStream_t stream) {
  (void)in_sizes; (void)n_in; (void)out_size; (void)ws_size;
  const float* x   = (const float*)d_in[0];
  const float* W1  = (const float*)d_in[1];
  const float* U1w = (const float*)d_in[2];
  const float* b1  = (const float*)d_in[3];
  const float* W2  = (const float*)d_in[4];
  const float* U2w = (const float*)d_in[5];
  const float* b2  = (const float*)d_in[6];
  const float* Wd1 = (const float*)d_in[7];
  const float* bd1 = (const float*)d_in[8];
  const float* Wd2 = (const float*)d_in[9];
  const float* bd2 = (const float*)d_in[10];
  const float* Wo  = (const float*)d_in[11];
  const float* bo  = (const float*)d_in[12];
  float* out = (float*)d_out;

  char* ws = (char*)d_ws;
  int*    syncp  = (int*)ws;
  __bf16* h1base = (__bf16*)(ws + WS_H1_OFF);
  __bf16* h2base = (__bf16*)(ws + WS_H2_OFF);
  __bf16* xbf    = (__bf16*)(ws + WS_X_OFF);

  // zero sync + h-state double buffers (poisoned 0xAA otherwise)
  hipMemsetAsync(d_ws, 0, WS_ZERO_BYTES, stream);

  // stage x as time-major bf16
  cvt_x_kernel<<<(TSTEPS * BSZ * FDIM) / 256, 256, 0, stream>>>(x, xbf);

  (void)hipFuncSetAttribute((const void*)lstm_persistent_kernel,
                            hipFuncAttributeMaxDynamicSharedMemorySize, SMEM_BYTES);
  lstm_persistent_kernel<<<NWG, 256, SMEM_BYTES, stream>>>(
      xbf, W1, U1w, b1, W2, U2w, b2, h1base, h2base, syncp);

  // final h2 (step 511) lives in h2 buffer 0
  head_kernel<<<1, 256, 0, stream>>>(h2base, Wd1, bd1, Wd2, bd2, Wo, bo, out);
}
